// SelfAttentionQuantBrevitas_56530359550416
// MI455X (gfx1250) — 8-bit matrix operand order repaired and checked by reading, compiled, not run on hardware
//
#include <hip/hip_runtime.h>
#include <hip/hip_bf16.h>
#include <math.h>


#define NROWS 31744
#define CDIM  1024
#define SEQ   31
#define NHEAD 16
#define HDIM  64
#define BN_EPS 1e-5f

#define AS1 __attribute__((address_space(1)))
#define AS3 __attribute__((address_space(3)))

typedef __attribute__((ext_vector_type(16))) int      i32x16;
typedef __attribute__((ext_vector_type(8)))  int      i32x8;
typedef __attribute__((ext_vector_type(4)))  int      i32x4;
typedef __attribute__((ext_vector_type(8)))  float    f32x8;
typedef __attribute__((ext_vector_type(16))) _Float16 f16x16;

#if __has_builtin(__builtin_amdgcn_cvt_pk_fp8_f32) && __has_builtin(__builtin_amdgcn_cvt_f32_fp8)
#define HW_FP8 1
#endif

#if __has_builtin(__builtin_amdgcn_global_load_async_to_lds_b128) && \
    __has_builtin(__builtin_amdgcn_s_wait_asynccnt)
#define HAS_ASYNC_LDS 1
#endif

__device__ __forceinline__ unsigned char f32_to_e4m3_sw(float x) {
    if (x != x) return 0x7F;
    unsigned s = (__float_as_uint(x) >> 31) << 7;
    float ax = fabsf(x);
    if (ax >= 448.0f) return (unsigned char)(s | 0x7E);
    if (ax == 0.0f)   return (unsigned char)s;
    int e; (void)frexpf(ax, &e);
    int ulp_e = (e >= -5) ? (e - 4) : -9;
    float q = ldexpf(rintf(ldexpf(ax, -ulp_e)), ulp_e);
    if (q == 0.0f)   return (unsigned char)s;
    if (q >= 448.0f) return (unsigned char)(s | 0x7E);
    int e2; float f2 = frexpf(q, &e2);
    if (e2 - 1 < -6) {
        int M = (int)rintf(ldexpf(q, 9));
        return (unsigned char)(s | (M & 7));
    }
    int E = e2 - 1 + 7;
    int M = (int)rintf(ldexpf(f2, 4)) - 8;
    if (M >= 8) { M = 0; E += 1; }
    if (E >= 16) return (unsigned char)(s | 0x7E);
    return (unsigned char)(s | (E << 3) | (M & 7));
}

__device__ __forceinline__ float e4m3_to_f32_sw(unsigned char b) {
    unsigned s = b >> 7;
    int E = (b >> 3) & 0xF, M = b & 7;
    float v;
    if ((b & 0x7F) == 0x7F) v = __uint_as_float(0x7FC00000u);
    else if (E == 0)        v = (float)M * 0.001953125f;
    else                    v = ldexpf(8.0f + (float)M, E - 10);
    return s ? -v : v;
}

__device__ __forceinline__ unsigned char f32_to_e4m3(float x) {
#ifdef HW_FP8
    return (unsigned char)(__builtin_amdgcn_cvt_pk_fp8_f32(x, x, 0, false) & 0xFF);
#else
    return f32_to_e4m3_sw(x);
#endif
}

__device__ __forceinline__ unsigned f32x4_to_e4m3x4(float a, float b, float c, float d) {
#ifdef HW_FP8
    int lo = __builtin_amdgcn_cvt_pk_fp8_f32(a, b, 0, false);
    int hi = __builtin_amdgcn_cvt_pk_fp8_f32(c, d, 0, false);
    return (unsigned)(lo & 0xFFFF) | ((unsigned)hi << 16);
#else
    return (unsigned)f32_to_e4m3_sw(a) | ((unsigned)f32_to_e4m3_sw(b) << 8) |
           ((unsigned)f32_to_e4m3_sw(c) << 16) | ((unsigned)f32_to_e4m3_sw(d) << 24);
#endif
}

__device__ __forceinline__ float e4m3_from_dword(unsigned u, int i) {
#ifdef HW_FP8
    switch (i & 3) {
    case 0:  return __builtin_amdgcn_cvt_f32_fp8((int)u, 0);
    case 1:  return __builtin_amdgcn_cvt_f32_fp8((int)u, 1);
    case 2:  return __builtin_amdgcn_cvt_f32_fp8((int)u, 2);
    default: return __builtin_amdgcn_cvt_f32_fp8((int)u, 3);
    }
#else
    return e4m3_to_f32_sw((unsigned char)((u >> (8 * (i & 3))) & 0xFF));
#endif
}

__device__ __forceinline__ float quant_dequant_e4m3(float x) {
#ifdef HW_FP8
    return __builtin_amdgcn_cvt_f32_fp8(
        __builtin_amdgcn_cvt_pk_fp8_f32(x, x, 0, false) & 0xFF, 0);
#else
    return e4m3_to_f32_sw(f32_to_e4m3_sw(x));
#endif
}

__global__ __launch_bounds__(256) void bn1_quant_kernel(
    const float* __restrict__ x,
    const float* __restrict__ g, const float* __restrict__ bb,
    const float* __restrict__ m, const float* __restrict__ var,
    unsigned* __restrict__ xq32)
{
    size_t i4 = (size_t)blockIdx.x * blockDim.x + threadIdx.x;
    size_t j  = i4 * 4;
    int c = (int)(j & (CDIM - 1));
    const float4 xv = *(const float4*)(x + j);
    float xn[4];
    #pragma unroll
    for (int t = 0; t < 4; ++t) {
        int cc = c + t;
        float sc = g[cc] * rsqrtf(var[cc] + BN_EPS);
        float xe = (t == 0) ? xv.x : (t == 1) ? xv.y : (t == 2) ? xv.z : xv.w;
        xn[t] = (xe - m[cc]) * sc + bb[cc];
    }
    xq32[i4] = f32x4_to_e4m3x4(xn[0], xn[1], xn[2], xn[3]);
}

__global__ __launch_bounds__(256) void weight_quant_kernel(
    const float* __restrict__ w, unsigned* __restrict__ w32)
{
    size_t i4 = (size_t)blockIdx.x * blockDim.x + threadIdx.x;
    const float4 wv = *(const float4*)(w + i4 * 4);
    w32[i4] = f32x4_to_e4m3x4(wv.x, wv.y, wv.z, wv.w);
}

__global__ __launch_bounds__(128) void gemm_fp8_kernel(
    const unsigned char* __restrict__ A8,
    const unsigned char* __restrict__ W8,
    unsigned char* __restrict__ out8,
    float* __restrict__ outf,
    const float* __restrict__ xres,
    int mode)
{
    const int lane  = threadIdx.x & 31;
    const int wid   = threadIdx.x >> 5;
    const int wlid  = blockIdx.x * 4 + wid;
    const int mtile = wlid >> 4;
    const int nstrip= wlid & 15;
    const int m0 = mtile * 16;
    const int n0 = nstrip * 64;
    const int mr = lane & 15;
    const int hi = lane >> 4;

    f32x8 acc[4] = {};

#ifdef HAS_ASYNC_LDS
    __shared__ __align__(16) unsigned char Atile[2][16][128];
    const int t = threadIdx.x;
    const unsigned char* gsrc = A8 + (size_t)(m0 + (t >> 3)) * CDIM + (t & 7) * 16;
    {
        __builtin_amdgcn_global_load_async_to_lds_b128(
            (AS1 i32x4*)(gsrc + 0 * 128),
            (AS3 i32x4*)&Atile[0][t >> 3][(t & 7) * 16], 0, 0);
        __builtin_amdgcn_global_load_async_to_lds_b128(
            (AS1 i32x4*)(gsrc + 1 * 128),
            (AS3 i32x4*)&Atile[1][t >> 3][(t & 7) * 16], 0, 0);
    }
#else
    const unsigned char* arow = A8 + (size_t)(m0 + mr) * CDIM;
#endif

    for (int kc = 0; kc < 8; ++kc) {
        const int k0 = kc * 128;
        union { i32x16 v; unsigned long long d[8]; } a;

#ifdef HAS_ASYNC_LDS
        if (kc < 7) __builtin_amdgcn_s_wait_asynccnt(1);
        else        __builtin_amdgcn_s_wait_asynccnt(0);
        __syncthreads();
        #pragma unroll
        for (int sb = 0; sb < 2; ++sb)
            #pragma unroll
            for (int j = 0; j < 4; ++j)
                a.d[sb * 4 + j] = *(const unsigned long long*)
                    &Atile[kc & 1][mr][sb * 64 + (j >> 1) * 32 + hi * 16 + (j & 1) * 8];
#else
        #pragma unroll
        for (int sb = 0; sb < 2; ++sb)
            #pragma unroll
            for (int j = 0; j < 4; ++j)
                a.d[sb * 4 + j] = *(const unsigned long long*)
                    (arow + k0 + sb * 64 + (j >> 1) * 32 + hi * 16 + (j & 1) * 8);
#endif

        #pragma unroll
        for (int nt = 0; nt < 4; ++nt) {
            union { i32x16 v; uint4 q[4]; } bm;
            const unsigned char* brow = W8 + (size_t)(n0 + nt * 16 + mr) * CDIM + k0;
            if (nt == 0 && kc + 1 < 8) __builtin_prefetch(brow + 128, 0, 1);
            #pragma unroll
            for (int c = 0; c < 4; ++c)
                bm.q[c] = *(const uint4*)(brow + c * 32 + hi * 16);
            acc[nt] = __builtin_amdgcn_wmma_f32_16x16x128_fp8_fp8(
                a.v, bm.v, (short)0, acc[nt], false, false);
        }

#ifdef HAS_ASYNC_LDS
        __syncthreads();
        if (kc + 2 < 8)
            __builtin_amdgcn_global_load_async_to_lds_b128(
                (AS1 i32x4*)(gsrc + (kc + 2) * 128),
                (AS3 i32x4*)&Atile[kc & 1][t >> 3][(t & 7) * 16], 0, 0);
#endif
    }

    #pragma unroll
    for (int nt = 0; nt < 4; ++nt) {
        #pragma unroll
        for (int r = 0; r < 8; ++r) {
            int row = m0 + r + 8 * hi;
            int col = n0 + nt * 16 + mr;
            float v = acc[nt][r];
            size_t o = (size_t)row * CDIM + col;
            if (mode == 0) out8[o] = f32_to_e4m3(v);
            else           outf[o] = quant_dequant_e4m3(v) + xres[o];
        }
    }
}

__global__ __launch_bounds__(128) void attention_kernel(
    const unsigned char* __restrict__ q8,
    const unsigned char* __restrict__ k8,
    const unsigned char* __restrict__ v8,
    const float* __restrict__ g2, const float* __restrict__ b2,
    const float* __restrict__ m2, const float* __restrict__ v2,
    unsigned char* __restrict__ att8)
{
    __shared__ __align__(16) float    eLDS[4][32][32];
    __shared__ __align__(16) _Float16 aLDS[4][32][32];
    __shared__ __align__(16) _Float16 vT  [4][64][32];

    const int lane = threadIdx.x & 31;
    const int wid  = threadIdx.x >> 5;
    const int bh   = blockIdx.x * 4 + wid;
    const int b    = bh >> 4;
    const int h    = bh & 15;
    const int mr   = lane & 15;
    const int hi   = lane >> 4;
    const size_t base = (size_t)b * SEQ * CDIM + (size_t)h * HDIM;

    union A64 { i32x8 v; unsigned long long d[4]; };
    union B64 { i32x8 v; uint4 q[2]; };
    A64 qa[2]; B64 kb[2];
    #pragma unroll
    for (int mt = 0; mt < 2; ++mt) {
        int s = mt * 16 + mr;
        #pragma unroll
        for (int j = 0; j < 4; ++j)
            qa[mt].d[j] = (s < SEQ)
                ? *(const unsigned long long*)(q8 + base + (size_t)s * CDIM + (j >> 1) * 32 + hi * 16 + (j & 1) * 8)
                : 0ull;
    }
    #pragma unroll
    for (int nt = 0; nt < 2; ++nt) {
        int s = nt * 16 + mr;
        #pragma unroll
        for (int c = 0; c < 2; ++c)
            kb[nt].q[c] = (s < SEQ)
                ? *(const uint4*)(k8 + base + (size_t)s * CDIM + c * 32 + hi * 16)
                : make_uint4(0u, 0u, 0u, 0u);
    }

    #pragma unroll
    for (int mt = 0; mt < 2; ++mt) {
        #pragma unroll
        for (int nt = 0; nt < 2; ++nt) {
            f32x8 e = {};
            e = __builtin_amdgcn_wmma_f32_16x16x64_fp8_fp8(
                qa[mt].v, kb[nt].v, (short)0, e, false, false);
            #pragma unroll
            for (int r = 0; r < 8; ++r) {
                int qs = mt * 16 + r + 8 * hi;
                int ks = nt * 16 + mr;
                float val = -3.0e38f;
                if (qs < SEQ && ks < SEQ) {
                    int idx = qs * SEQ + ks;
                    float sc = g2[idx] * rsqrtf(v2[idx] + BN_EPS);
                    val = (e[r] - m2[idx]) * sc + b2[idx];
                }
                eLDS[wid][qs][ks] = val;
            }
        }
    }
    __syncthreads();

    {
        int row = lane;
        if (row < SEQ) {
            float mx = -3.0e38f;
            for (int j = 0; j < SEQ; ++j) mx = fmaxf(mx, eLDS[wid][row][j]);
            float sum = 0.f;
            for (int j = 0; j < SEQ; ++j) sum += __expf(eLDS[wid][row][j] - mx);
            float inv = 1.0f / sum;
            for (int j = 0; j < SEQ; ++j)
                aLDS[wid][row][j] = (_Float16)(__expf(eLDS[wid][row][j] - mx) * inv);
            aLDS[wid][row][SEQ] = (_Float16)0.f;
        } else {
            for (int j = 0; j < 32; ++j) aLDS[wid][row][j] = (_Float16)0.f;
        }
        int l = lane;
        #pragma unroll 4
        for (int dw = 0; dw < 16; ++dw) {
            unsigned u = (l < SEQ)
                ? *(const unsigned*)(v8 + base + (size_t)l * CDIM + dw * 4) : 0u;
            #pragma unroll
            for (int bsel = 0; bsel < 4; ++bsel) {
                _Float16 hv = (l < SEQ) ? (_Float16)e4m3_from_dword(u, bsel)
                                        : (_Float16)0.f;
                vT[wid][dw * 4 + bsel][l] = hv;
            }
        }
    }
    __syncthreads();

    union H16 { f16x16 v; uint4 q[2]; };
    #pragma unroll
    for (int mt = 0; mt < 2; ++mt) {
        H16 ah;
        const char* ar = (const char*)&aLDS[wid][mt * 16 + mr][0];
        ah.q[0] = *(const uint4*)(ar + 16 * hi);
        ah.q[1] = *(const uint4*)(ar + 32 + 16 * hi);
        #pragma unroll
        for (int nt = 0; nt < 4; ++nt) {
            H16 bm;
            const char* br = (const char*)&vT[wid][nt * 16 + mr][0];
            bm.q[0] = *(const uint4*)(br + 16 * hi);
            bm.q[1] = *(const uint4*)(br + 32 + 16 * hi);
            f32x8 o = {};
            o = __builtin_amdgcn_wmma_f32_16x16x32_f16(
                false, ah.v, false, bm.v, (short)0, o, false, false);
            #pragma unroll
            for (int r = 0; r < 8; ++r) {
                int qs = mt * 16 + r + 8 * hi;
                int dc = nt * 16 + mr;
                if (qs < SEQ)
                    att8[base + (size_t)qs * CDIM + dc] = f32_to_e4m3(o[r]);
            }
        }
    }
}

extern "C" void kernel_launch(void* const* d_in, const int* in_sizes, int n_in,
                              void* d_out, int out_size, void* d_ws, size_t ws_size,
                              hipStream_t stream) {
    const float* x  = (const float*)d_in[0];
    const float* Wq = (const float*)d_in[1];
    const float* Wk = (const float*)d_in[2];
    const float* Wv = (const float*)d_in[3];
    const float* Wo = (const float*)d_in[4];
    const float* g1 = (const float*)d_in[5];
    const float* b1 = (const float*)d_in[6];
    const float* m1 = (const float*)d_in[7];
    const float* v1 = (const float*)d_in[8];
    const float* g2 = (const float*)d_in[9];
    const float* b2 = (const float*)d_in[10];
    const float* m2 = (const float*)d_in[11];
    const float* v2 = (const float*)d_in[12];
    float* out = (float*)d_out;

    const size_t NC = (size_t)NROWS * CDIM;
    const size_t WW = (size_t)CDIM * CDIM;
    unsigned char* ws = (unsigned char*)d_ws;
    size_t off = 0;
    auto take = [&](size_t bytes) {
        unsigned char* p = ws + off;
        off = (off + bytes + 255) & ~(size_t)255;
        return p;
    };
    unsigned char* xq8  = take(NC);
    unsigned char* wq8  = take(WW);
    unsigned char* wk8  = take(WW);
    unsigned char* wv8  = take(WW);
    unsigned char* wo8  = take(WW);
    unsigned char* q8   = take(NC);
    unsigned char* k8   = take(NC);
    unsigned char* vv8  = take(NC);
    unsigned char* att8 = take(NC);

    bn1_quant_kernel<<<(unsigned)(NC / (256 * 4)), 256, 0, stream>>>(
        x, g1, b1, m1, v1, (unsigned*)xq8);

    weight_quant_kernel<<<(unsigned)(WW / (256 * 4)), 256, 0, stream>>>(Wq, (unsigned*)wq8);
    weight_quant_kernel<<<(unsigned)(WW / (256 * 4)), 256, 0, stream>>>(Wk, (unsigned*)wk8);
    weight_quant_kernel<<<(unsigned)(WW / (256 * 4)), 256, 0, stream>>>(Wv, (unsigned*)wv8);
    weight_quant_kernel<<<(unsigned)(WW / (256 * 4)), 256, 0, stream>>>(Wo, (unsigned*)wo8);

    const unsigned gemmGrid = (NROWS / 16) * 16 / 4;
    gemm_fp8_kernel<<<gemmGrid, 128, 0, stream>>>(xq8, wq8, q8, nullptr, nullptr, 0);
    gemm_fp8_kernel<<<gemmGrid, 128, 0, stream>>>(xq8, wk8, k8, nullptr, nullptr, 0);
    gemm_fp8_kernel<<<gemmGrid, 128, 0, stream>>>(xq8, wv8, vv8, nullptr, nullptr, 0);

    attention_kernel<<<(1024 * NHEAD) / 4, 128, 0, stream>>>(
        q8, k8, vv8, g2, b2, m2, v2, att8);

    gemm_fp8_kernel<<<gemmGrid, 128, 0, stream>>>(att8, wo8, nullptr, out, x, 1);
}
